// PartBasedGraphConv_17454747090956
// MI455X (gfx1250) — matrix operand order repaired and checked by reading, not compiled, not run on hardware
//
#include <hip/hip_runtime.h>
#include <stdint.h>

typedef __bf16 bf16_t;
typedef __attribute__((ext_vector_type(16))) __bf16   v16bf;
typedef __attribute__((ext_vector_type(8)))  float    v8f;
typedef __attribute__((ext_vector_type(4)))  float    v4f;
typedef __attribute__((ext_vector_type(4)))  uint32_t v4u;

static constexpr int NB   = 8192;   // batch
static constexpr int NJ   = 17;     // joints
static constexpr int NP   = 5;      // parts
static constexpr int DIM  = 512;
static constexpr int MROWS = NB * NP;          // 40960 GEMM rows
static constexpr size_t WT_ELEMS = 3ull * DIM * DIM;       // bf16
static constexpr size_t A_ELEMS  = (size_t)MROWS * DIM;    // bf16 per cheb term

// cheb[k] for the fixed part graph (L = I - A_rownorm since rows sum to 1; T2 = 2L^2 - I)
__device__ __constant__ float CHEB[3][5][5] = {
  {{1,0,0,0,0},{0,1,0,0,0},{0,0,1,0,0},{0,0,0,1,0},{0,0,0,0,1}},
  {{ 0.5f,0.0f,-0.5f,0.0f,0.0f},
   { 0.0f,0.5f,-0.5f,0.0f,0.0f},
   {-0.2f,-0.2f,0.8f,-0.2f,-0.2f},
   { 0.0f,0.0f,-0.5f,0.5f,0.0f},
   { 0.0f,0.0f,-0.5f,0.0f,0.5f}},
  {{-0.30f, 0.20f,-1.30f, 0.20f, 0.20f},
   { 0.20f,-0.30f,-1.30f, 0.20f, 0.20f},
   {-0.52f,-0.52f, 1.08f,-0.52f,-0.52f},
   { 0.20f, 0.20f,-1.30f,-0.30f, 0.20f},
   { 0.20f, 0.20f,-1.30f, 0.20f,-0.30f}}
};

__device__ __constant__ int   JPART[17] = {2,0,0,0,1,1,1,2,2,2,2,3,3,3,4,4,4};
__device__ __constant__ float PINV[5]   = {1.f/3.f, 1.f/3.f, 0.2f, 1.f/3.f, 1.f/3.f};
__device__ __constant__ int   PCNT[5]   = {3,3,5,3,3};
__device__ __constant__ int   PJOINT[5][5] = {{1,2,3,0,0},{4,5,6,0,0},{0,7,8,9,10},{11,12,13,0,0},{14,15,16,0,0}};

__device__ inline uint16_t f2bf(float f) {
  uint32_t u = __builtin_bit_cast(uint32_t, f);
  uint32_t r = u + 0x7FFFu + ((u >> 16) & 1u);   // round-to-nearest-even
  return (uint16_t)(r >> 16);
}

// Guaranteed global (addrspace 1) 16-byte load — keeps the compiler off the
// FLAT path (flat loads tie up DScnt + the LDS port on CDNA5). Uses an
// ext_vector type (plain builtin vector) so the addrspace(1) deref is legal.
__device__ inline v4u gload_u4(const uint16_t* p) {
  typedef __attribute__((address_space(1))) const v4u GV;
  return *(GV*)(unsigned long long)(p);
}

// ---------------- Kernel 1: W[k][d][n] (f32) -> Wt[k][n][d] (bf16) ----------------
__global__ __launch_bounds__(256) void wprep_k(const float* __restrict__ W,
                                               uint16_t* __restrict__ Wt) {
  int idx = blockIdx.x * 256 + threadIdx.x;            // (k,n,d), d fastest
  int k = idx >> 18;
  int n = (idx >> 9) & 511;
  int d = idx & 511;
  Wt[idx] = f2bf(W[((size_t)k * DIM + d) * DIM + n]);
}

// ---------------- Kernel 2: fused part-mean pool + cheb mixing -> 3 bf16 A mats ----
__global__ __launch_bounds__(256) void pool_cheb_k(const float* __restrict__ x,
                                                   uint16_t* __restrict__ A) {
  int idx = blockIdx.x * 256 + threadIdx.x;            // NB * 128 threads
  int b  = idx >> 7;
  int c4 = (idx & 127) << 2;                           // 4 consecutive columns
  const float* xb = x + ((size_t)b * NJ) * DIM + c4;

  float f[5][4];
  #pragma unroll
  for (int m = 0; m < 5; ++m) { f[m][0]=f[m][1]=f[m][2]=f[m][3]=0.f; }

  #pragma unroll
  for (int j = 0; j < NJ; ++j) {
    // x is read exactly once -> non-temporal (ext_vector type: HIP float4 is
    // rejected by the nontemporal builtin)
    v4f v = __builtin_nontemporal_load((const v4f*)(xb + (size_t)j * DIM));
    int p = JPART[j];
    float w = PINV[p];
    f[p][0] += w * v[0]; f[p][1] += w * v[1]; f[p][2] += w * v[2]; f[p][3] += w * v[3];
  }

  #pragma unroll
  for (int k = 0; k < 3; ++k) {
    #pragma unroll
    for (int n = 0; n < NP; ++n) {
      float o0=0.f,o1=0.f,o2=0.f,o3=0.f;
      #pragma unroll
      for (int m = 0; m < NP; ++m) {
        float cm = CHEB[k][n][m];
        o0 += cm*f[m][0]; o1 += cm*f[m][1]; o2 += cm*f[m][2]; o3 += cm*f[m][3];
      }
      uint2 pk;
      pk.x = (uint32_t)f2bf(o0) | ((uint32_t)f2bf(o1) << 16);
      pk.y = (uint32_t)f2bf(o2) | ((uint32_t)f2bf(o3) << 16);
      // A matrices are re-read by the GEMM -> regular (cacheable) store
      *(uint2*)(A + (size_t)k * A_ELEMS + ((size_t)(b * NP + n) * DIM + c4)) = pk;
    }
  }
}

// ---------------- Kernel 3: bf16 WMMA GEMM + bias + scatter to joints --------------
// grid = (4 n-blocks, 320 m-blocks), 256 threads = 8 waves.
// Workgroup tile: 128 rows x 128 cols. Wave = 2 m-tiles x 4 n-tiles of 16x16.
__global__ __launch_bounds__(256) void gemm_scatter_k(const uint16_t* __restrict__ Wt,
                                                      const uint16_t* __restrict__ A,
                                                      const float* __restrict__ bias,
                                                      float* __restrict__ out) {
  const int lane = threadIdx.x & 31;
  const int wave = threadIdx.x >> 5;
  const int wm = wave & 3;            // 4 waves along M
  const int wn = wave >> 2;           // 2 waves along N
  const int row0 = blockIdx.y * 128 + wm * 32;   // 2 m-tiles
  const int col0 = blockIdx.x * 128 + wn * 64;   // 4 n-tiles
  const int lr = lane & 15;
  const int lh = lane >> 4;

  v8f acc[2][4];
  #pragma unroll
  for (int mt = 0; mt < 2; ++mt)
    #pragma unroll
    for (int nt = 0; nt < 4; ++nt)
      acc[mt][nt] = {};

  union Pack32 { v4u q[2]; v16bf v; };

  #pragma unroll 1
  for (int k3 = 0; k3 < 3; ++k3) {
    const uint16_t* Ak = A + (size_t)k3 * A_ELEMS;
    const uint16_t* Bw = Wt + (size_t)k3 * DIM * DIM;
    #pragma unroll 2
    for (int kk = 0; kk < DIM; kk += 32) {
      v16bf a[2], b[4];
      // A operand: lane lr = row within tile; elems 0-7 = K[kk+lh*8 ..], elems 8-15 = K[kk+16+lh*8 ..]
      #pragma unroll
      for (int mt = 0; mt < 2; ++mt) {
        const uint16_t* ap = Ak + (size_t)(row0 + mt * 16 + lr) * DIM + kk + lh * 8;
        Pack32 t;
        t.q[0] = gload_u4(ap);
        t.q[1] = gload_u4(ap + 16);
        a[mt] = t.v;
      }
      // B operand: lane lr = output col; elems 0-7 = K[kk+lh*8 ..], elems 8-15 = K[kk+16+lh*8 ..] from Wt[n][k]
      #pragma unroll
      for (int nt = 0; nt < 4; ++nt) {
        const uint16_t* bp = Bw + (size_t)(col0 + nt * 16 + lr) * DIM + kk + lh * 8;
        Pack32 t;
        t.q[0] = gload_u4(bp);
        t.q[1] = gload_u4(bp + 16);
        b[nt] = t.v;
      }
      #pragma unroll
      for (int mt = 0; mt < 2; ++mt)
        #pragma unroll
        for (int nt = 0; nt < 4; ++nt)
          acc[mt][nt] = __builtin_amdgcn_wmma_f32_16x16x32_bf16(
              false, a[mt], false, b[nt], (short)0, acc[mt][nt], false, false);
    }
  }

  // Epilogue: C layout — VGPR r: lanes0-15 -> M=r, lanes16-31 -> M=8+r; N = lr.
  // Output is write-once, never re-read by us -> non-temporal stores keep L2
  // free for A-matrix reuse across the 4 n-blocks.
  #pragma unroll
  for (int mt = 0; mt < 2; ++mt) {
    #pragma unroll
    for (int nt = 0; nt < 4; ++nt) {
      const int col = col0 + nt * 16 + lr;
      const float bv = bias[col];
      #pragma unroll
      for (int r = 0; r < 8; ++r) {
        const int grow = row0 + mt * 16 + lh * 8 + r;  // global A-row = b*5 + part
        const int bidx = grow / 5;
        const int p    = grow - bidx * 5;
        const float val = acc[mt][nt][r] + bv;
        const int cnt = PCNT[p];
        #pragma unroll 5
        for (int t = 0; t < 5; ++t) {
          if (t < cnt) {
            const int j = PJOINT[p][t];
            __builtin_nontemporal_store(val, out + ((size_t)bidx * NJ + j) * DIM + col);
          }
        }
      }
    }
  }
}

extern "C" void kernel_launch(void* const* d_in, const int* in_sizes, int n_in,
                              void* d_out, int out_size, void* d_ws, size_t ws_size,
                              hipStream_t stream) {
  const float* x    = (const float*)d_in[0];   // [8192,17,512]
  const float* W    = (const float*)d_in[1];   // [3,1,512,512]
  const float* bias = (const float*)d_in[2];   // [512]
  float* out = (float*)d_out;                  // [8192,17,512]

  uint16_t* Wt = (uint16_t*)d_ws;
  uint16_t* A  = Wt + WT_ELEMS;                // 3 contiguous bf16 A-matrices

  // 1) weight transpose + bf16 cast: 3*512*512 elements
  wprep_k<<<(int)(WT_ELEMS / 256), 256, 0, stream>>>(W, Wt);

  // 2) fused pool + cheb: NB*128 threads (each handles 4 columns)
  pool_cheb_k<<<NB * 128 / 256, 256, 0, stream>>>(x, A);

  // 3) GEMM + bias + scatter: 128x128 tiles over [40960 x 512]
  dim3 grid(DIM / 128, MROWS / 128);           // (4, 320); x fastest for L2 A-reuse
  gemm_scatter_k<<<grid, 256, 0, stream>>>(Wt, A, bias, out);
}
